// MFABlock_74680891343375
// MI455X (gfx1250) — matrix operand order repaired and checked by reading, not compiled, not run on hardware
//
#include <hip/hip_runtime.h>
#include <stdint.h>

#define L_SEQ   4096
#define NTOK    8192        // B_SZ * L_SEQ

typedef __attribute__((ext_vector_type(16))) __bf16 bf16x16;
typedef __attribute__((ext_vector_type(8)))  float  floatx8;

union BF16Frag { bf16x16 v; unsigned short u[16]; };

__device__ __forceinline__ unsigned short f2bf_u(float f) {
    unsigned u = __builtin_bit_cast(unsigned, f);
    unsigned r = u + 0x7FFFu + ((u >> 16) & 1u);     // round to nearest even
    return (unsigned short)(r >> 16);
}

__device__ __forceinline__ floatx8 zero8() {
    floatx8 z = {0.f,0.f,0.f,0.f,0.f,0.f,0.f,0.f};
    return z;
}
__device__ __forceinline__ bf16x16 zfrag() {
    BF16Frag f;
    #pragma unroll
    for (int j = 0; j < 16; ++j) f.u[j] = 0;
    return f.v;
}

// ---- WMMA fragment loaders (wave32, 16x16x32 bf16) -------------------------
// A (16x32, 16-bit): lane lo = row m; halves K = kb + hi*16 + 0..15 (contiguous)
__device__ __forceinline__ bf16x16 load_a_bf16(const unsigned short* row, int kb, int hi) {
    BF16Frag f;
    const unsigned short* p0 = row + kb + hi * 16;
    const unsigned short* p1 = row + kb + 8 + hi * 16;
    #pragma unroll
    for (int j = 0; j < 8; ++j) { f.u[j] = p0[j]; f.u[8 + j] = p1[j]; }
    return f.v;
}
__device__ __forceinline__ bf16x16 load_a_f32(const float* row, int kb, int hi) {
    BF16Frag f;
    const float* p0 = row + kb + hi * 16;
    const float* p1 = row + kb + 8 + hi * 16;
    #pragma unroll
    for (int j = 0; j < 8; ++j) { f.u[j] = f2bf_u(p0[j]); f.u[8 + j] = f2bf_u(p1[j]); }
    return f.v;
}
// B (32x16, 16-bit): lane lo = column n; halves K = kb + hi*16 + 0..15 (contiguous)
__device__ __forceinline__ bf16x16 load_b_bf16(const unsigned short* rown, int kb, int hi) {
    BF16Frag f;
    const unsigned short* p = rown + kb + hi * 16;
    #pragma unroll
    for (int j = 0; j < 16; ++j) f.u[j] = p[j];
    return f.v;
}
__device__ __forceinline__ bf16x16 load_b_f32(const float* rown, int kb, int hi) {
    BF16Frag f;
    const float* p = rown + kb + hi * 16;
    #pragma unroll
    for (int j = 0; j < 16; ++j) f.u[j] = f2bf_u(p[j]);
    return f.v;
}
__device__ __forceinline__ floatx8 wmma_bf16(bf16x16 a, bf16x16 b, floatx8 c) {
    return __builtin_amdgcn_wmma_f32_16x16x32_bf16(false, a, false, b, (short)0, c, false, false);
}

// ============================================================================
// 0) weight repack to bf16 (once per launch; deterministic)
// ============================================================================
__global__ void k_pack(const float* ipw, const float* xwf, const float* xwb, const float* xws,
                       const float* opw, const float* f1w, const float* f2w,
                       unsigned short* wip, unsigned short* wxp, unsigned short* wop,
                       unsigned short* wf1, unsigned short* wf2) {
    int t = blockIdx.x * blockDim.x + threadIdx.x;
    int stride = gridDim.x * blockDim.x;
    for (int i = t; i < 512 * 128; i += stride) wip[i] = f2bf_u(ipw[i]);
    for (int i = t; i < 3 * 48 * 256; i += stride) {
        int br = i / (48 * 256); int r = (i / 256) % 48; int c = i % 256;
        const float* src = (br == 0) ? xwf : ((br == 1) ? xwb : xws);
        wxp[i] = (r < 40) ? f2bf_u(src[r * 256 + c]) : (unsigned short)0;
    }
    for (int i = t; i < 128 * 256; i += stride) wop[i] = f2bf_u(opw[i]);
    for (int i = t; i < 128 * 9 * 256; i += stride) {
        int c = i / (9 * 256); int tp = (i / 256) % 9; int ci = i % 256;
        wf1[i] = f2bf_u(f1w[(c * 256 + ci) * 9 + tp]);
        wf2[i] = f2bf_u(f2w[(c * 256 + ci) * 9 + tp]);
    }
}

// ============================================================================
// 1) layernorm over 128 channels -> bf16 tokens [NTOK][128]
// ============================================================================
__global__ void k_ln(const float* __restrict__ x, const float* lw, const float* lb,
                     unsigned short* __restrict__ xnb) {
    int tok = blockIdx.x; int c = threadIdx.x;
    int b = tok >> 12; int l = tok & 4095;
    float v = x[((size_t)b * 128 + c) * 4096 + l];
    float s = v, s2 = v * v;
    #pragma unroll
    for (int m = 16; m >= 1; m >>= 1) { s += __shfl_xor(s, m, 32); s2 += __shfl_xor(s2, m, 32); }
    __shared__ float rs[4], rq[4];
    int w = threadIdx.x >> 5;
    if ((threadIdx.x & 31) == 0) { rs[w] = s; rq[w] = s2; }
    __syncthreads();
    s = rs[0] + rs[1] + rs[2] + rs[3];
    s2 = rq[0] + rq[1] + rq[2] + rq[3];
    float mu = s * (1.f / 128.f);
    float var = s2 * (1.f / 128.f) - mu * mu;
    float rstd = rsqrtf(var + 1e-5f);
    xnb[(size_t)tok * 128 + c] = f2bf_u((v - mu) * rstd * lw[c] + lb[c]);
}

// ============================================================================
// 2) in_proj: [8192x128] x [512x128]^T -> xz f32 [8192x512]   (wave: 16x64)
// ============================================================================
__global__ void k_inproj(const unsigned short* __restrict__ xnb,
                         const unsigned short* __restrict__ wip,
                         float* __restrict__ xz) {
    int wave = (blockIdx.x * 256 + threadIdx.x) >> 5;
    int lane = threadIdx.x & 31, lo = lane & 15, hi = lane >> 4;
    int mt = wave >> 3, ng = wave & 7;                  // 512 M-tiles, 8 N-groups of 64
    const unsigned short* arow = xnb + (size_t)(mt * 16 + lo) * 128;
    floatx8 acc[4] = {zero8(), zero8(), zero8(), zero8()};
    #pragma unroll
    for (int kb = 0; kb < 128; kb += 32) {
        bf16x16 a = load_a_bf16(arow, kb, hi);
        #pragma unroll
        for (int nt = 0; nt < 4; ++nt) {
            int n = ng * 64 + nt * 16 + lo;
            acc[nt] = wmma_bf16(a, load_b_bf16(wip + (size_t)n * 128, kb, hi), acc[nt]);
        }
    }
    int mb = mt * 16 + hi * 8;
    #pragma unroll
    for (int nt = 0; nt < 4; ++nt) {
        int n = ng * 64 + nt * 16 + lo;
        #pragma unroll
        for (int r = 0; r < 8; ++r) xz[(size_t)(mb + r) * 512 + n] = acc[nt][r];
    }
}

// ============================================================================
// 3) causal depthwise conv1d + silu, per branch with its scan-order mapping
// ============================================================================
__global__ void k_conv1d(const float* __restrict__ xz,
                         const float* cwf, const float* cbf,
                         const float* cwb, const float* cbb,
                         const float* cws, const float* cbs,
                         float* __restrict__ uc) {
    int br = blockIdx.y;
    int tok = blockIdx.x; int b = tok >> 12; int l = tok & 4095;
    int d = threadIdx.x;
    const float* cw = (br == 0) ? cwf : ((br == 1) ? cwb : cws);
    const float* cb = (br == 0) ? cbf : ((br == 1) ? cbb : cbs);
    float acc = cb[d];
    #pragma unroll
    for (int t = 0; t < 4; ++t) {
        int j = l - 3 + t;
        if (j < 0) continue;
        int m = (br == 0) ? j : ((br == 1) ? (4095 - j) : ((j & 3) * 1024 + (j >> 2)));
        acc += cw[d * 4 + t] * xz[((size_t)(b * 4096 + m)) * 512 + d];
    }
    float sg = 1.f / (1.f + __expf(-acc));
    uc[(((size_t)br * 2 + b) * 4096 + l) * 256 + d] = acc * sg;
}

// ============================================================================
// 4) x_proj: uc[8192x256] x xw[48x256]^T -> proj f32 [8192x48]  (wave: 16x48)
// ============================================================================
__global__ void k_xproj(const float* __restrict__ uc,
                        const unsigned short* __restrict__ wxp,
                        float* __restrict__ proj) {
    int br = blockIdx.y;
    int mt = (blockIdx.x * 256 + threadIdx.x) >> 5;      // 512 M-tiles
    int lane = threadIdx.x & 31, lo = lane & 15, hi = lane >> 4;
    const float* arow = uc + (size_t)br * NTOK * 256 + (size_t)(mt * 16 + lo) * 256;
    const unsigned short* wb = wxp + br * 48 * 256;
    floatx8 acc[3] = {zero8(), zero8(), zero8()};
    #pragma unroll
    for (int kb = 0; kb < 256; kb += 32) {
        bf16x16 a = load_a_f32(arow, kb, hi);
        #pragma unroll
        for (int nt = 0; nt < 3; ++nt)
            acc[nt] = wmma_bf16(a, load_b_bf16(wb + (size_t)(nt * 16 + lo) * 256, kb, hi), acc[nt]);
    }
    float* pb = proj + (size_t)br * NTOK * 48;
    int mb = mt * 16 + hi * 8;
    #pragma unroll
    for (int nt = 0; nt < 3; ++nt) {
        int n = nt * 16 + lo;
        #pragma unroll
        for (int r = 0; r < 8; ++r) pb[(size_t)(mb + r) * 48 + n] = acc[nt][r];
    }
}

// ============================================================================
// 5) delta = softplus(dt @ dw^T + db)
// ============================================================================
__global__ void k_delta(const float* __restrict__ proj,
                        const float* dwf, const float* dbf,
                        const float* dwb, const float* dbb,
                        const float* dws, const float* dbs,
                        float* __restrict__ delta) {
    int br = blockIdx.y; int tok = blockIdx.x; int d = threadIdx.x;
    const float* dw = (br == 0) ? dwf : ((br == 1) ? dwb : dws);
    const float* db = (br == 0) ? dbf : ((br == 1) ? dbb : dbs);
    const float* pr = proj + ((size_t)br * NTOK + tok) * 48;
    float a = db[d];
    #pragma unroll
    for (int r = 0; r < 8; ++r) a += pr[r] * dw[d * 8 + r];
    float sp = (a > 20.f) ? a : log1pf(__expf(a));
    delta[((size_t)br * NTOK + tok) * 256 + d] = sp;
}

// ============================================================================
// 6) selective scan: one lane per (channel d, state n); shuffle-reduce over n.
//    Prefetch rows ~16 iterations ahead so the dependent h-chain never stalls.
// ============================================================================
__global__ void k_scan(const float* __restrict__ uc, const float* __restrict__ proj,
                       const float* __restrict__ delta,
                       const float* Alf, const float* Df,
                       const float* Alb, const float* Db,
                       const float* Als, const float* Ds,
                       float* __restrict__ yscan) {
    int gid = blockIdx.x;                  // 96 blocks
    int br = gid >> 5;
    int rem = gid & 31; int b = rem >> 4; int wg = rem & 15;
    int t = threadIdx.x;
    int d = wg * 16 + (t >> 4); int n = t & 15;
    const float* Al = (br == 0) ? Alf : ((br == 1) ? Alb : Als);
    const float* Dv = (br == 0) ? Df  : ((br == 1) ? Db  : Ds);
    float Aa = -__expf(Al[d * 16 + n]);
    float Dd = Dv[d];
    size_t base = ((size_t)br * 2 + b) * 4096;
    const float* ucb = uc + base * 256;
    const float* prb = proj + base * 48;
    const float* dlb = delta + base * 256;
    float* yb = yscan + base * 256;
    float h = 0.f;
    for (int l = 0; l < 4096; ++l) {
        if (l + 16 < 4096) {
            __builtin_prefetch(&dlb[(size_t)(l + 16) * 256 + d], 0, 1);
            __builtin_prefetch(&ucb[(size_t)(l + 16) * 256 + d], 0, 1);
            __builtin_prefetch(&prb[(l + 16) * 48 + n], 0, 1);
        }
        float dl = dlb[(size_t)l * 256 + d];
        float uv = ucb[(size_t)l * 256 + d];
        float bm = prb[l * 48 + 8 + n];
        float cm = prb[l * 48 + 24 + n];
        h = __expf(dl * Aa) * h + dl * uv * bm;
        float s = h * cm;
        s += __shfl_xor(s, 1, 32); s += __shfl_xor(s, 2, 32);
        s += __shfl_xor(s, 4, 32); s += __shfl_xor(s, 8, 32);
        if (n == 0) yb[(size_t)l * 256 + d] = s + uv * Dd;
    }
}

// ============================================================================
// 7) combine: reorder branches, gate with silu(z); emit bf16 q/k/v/ysum
// ============================================================================
__global__ void k_combine(const float* __restrict__ yscan, const float* __restrict__ xz,
                          unsigned short* __restrict__ ysum, unsigned short* __restrict__ qb,
                          unsigned short* __restrict__ kbuf, unsigned short* __restrict__ vb) {
    int tok = blockIdx.x; int b = tok >> 12; int l = tok & 4095; int d = threadIdx.x;
    size_t nb = (size_t)NTOK * 256;
    size_t bo = (size_t)b * 4096;
    float yf = yscan[(bo + l) * 256 + d];
    float yr = yscan[nb + (bo + (4095 - l)) * 256 + d];
    int inv = (l & 1023) * 4 + (l >> 10);
    float ys = yscan[2 * nb + (bo + inv) * 256 + d];
    float z = xz[(size_t)tok * 512 + 256 + d];
    float g = z / (1.f + __expf(-z));
    size_t o = (size_t)tok * 256 + d;
    ysum[o] = f2bf_u((yf + yr + ys) * g);
    qb[o] = f2bf_u(yf); kbuf[o] = f2bf_u(yr); vb[o] = f2bf_u(ys);
}

// ============================================================================
// 8) out_proj -> bf16 NHWC into cat channels [128..255]   (wave: 16x64)
// ============================================================================
__global__ void k_outproj(const unsigned short* __restrict__ ysum,
                          const unsigned short* __restrict__ wop,
                          unsigned short* __restrict__ cat) {
    int wave = (blockIdx.x * 256 + threadIdx.x) >> 5;
    int lane = threadIdx.x & 31, lo = lane & 15, hi = lane >> 4;
    int mt = wave >> 1, ng = wave & 1;
    const unsigned short* arow = ysum + (size_t)(mt * 16 + lo) * 256;
    floatx8 acc[4] = {zero8(), zero8(), zero8(), zero8()};
    #pragma unroll
    for (int kb = 0; kb < 256; kb += 32) {
        bf16x16 a = load_a_bf16(arow, kb, hi);
        #pragma unroll
        for (int nt = 0; nt < 4; ++nt) {
            int c = ng * 64 + nt * 16 + lo;
            acc[nt] = wmma_bf16(a, load_b_bf16(wop + (size_t)c * 256, kb, hi), acc[nt]);
        }
    }
    int mb = mt * 16 + hi * 8;
    #pragma unroll
    for (int nt = 0; nt < 4; ++nt) {
        int c = ng * 64 + nt * 16 + lo;
        #pragma unroll
        for (int r = 0; r < 8; ++r)
            cat[(size_t)(mb + r) * 256 + 128 + c] = f2bf_u(acc[nt][r]);
    }
}

// ============================================================================
// 9) G[b,d,e] = sum_l q[l,d] k[l,e]  — LDS-staged transpose GEMM (K=4096)
// ============================================================================
__global__ void k_qk(const unsigned short* __restrict__ qb,
                     const unsigned short* __restrict__ kbuf,
                     float* __restrict__ G) {
    int bid = blockIdx.x;                  // 32 blocks: b(2) x mtg(4) x ntg(4)
    int b = bid >> 4; int mtg = (bid >> 2) & 3; int ntg = bid & 3;
    __shared__ unsigned short qT[64 * 36];
    __shared__ unsigned short kT[64 * 36];
    int t = threadIdx.x, lane = t & 31, w = t >> 5, lo = lane & 15, hi = lane >> 4;
    int mt = w >> 1, npair = w & 1;
    floatx8 acc0 = zero8(), acc1 = zero8();
    const unsigned short* qbase = qb + (size_t)b * 4096 * 256;
    const unsigned short* kbase = kbuf + (size_t)b * 4096 * 256;
    int d0 = mtg * 64, e0 = ntg * 64;
    for (int kb0 = 0; kb0 < 4096; kb0 += 32) {
        __syncthreads();
        #pragma unroll
        for (int pass = 0; pass < 8; ++pass) {
            int e = pass * 256 + t;
            int ll = e >> 6, dd = e & 63;
            qT[dd * 36 + ll] = qbase[(size_t)(kb0 + ll) * 256 + d0 + dd];
            kT[dd * 36 + ll] = kbase[(size_t)(kb0 + ll) * 256 + e0 + dd];
            if (kb0 + 32 < 4096) {    // pull the next chunk toward the WGP cache
                __builtin_prefetch(&qbase[(size_t)(kb0 + 32 + ll) * 256 + d0 + dd], 0, 1);
                __builtin_prefetch(&kbase[(size_t)(kb0 + 32 + ll) * 256 + e0 + dd], 0, 1);
            }
        }
        __syncthreads();
        bf16x16 a  = load_a_bf16(qT + (mt * 16 + lo) * 36, 0, hi);
        bf16x16 b0 = load_b_bf16(kT + (npair * 32 + lo) * 36, 0, hi);
        bf16x16 b1 = load_b_bf16(kT + (npair * 32 + 16 + lo) * 36, 0, hi);
        acc0 = wmma_bf16(a, b0, acc0);
        acc1 = wmma_bf16(a, b1, acc1);
    }
    float* Gb = G + (size_t)b * 256 * 256;
    int mrow = d0 + mt * 16 + hi * 8;
    int nc = e0 + npair * 32 + lo;
    #pragma unroll
    for (int r = 0; r < 8; ++r) {
        Gb[(size_t)(mrow + r) * 256 + nc] = acc0[r];
        Gb[(size_t)(mrow + r) * 256 + nc + 16] = acc1[r];
    }
}

// ============================================================================
// 10) row softmax over 256 (one wave per row)
// ============================================================================
__global__ void k_softmax(const float* __restrict__ G, float* __restrict__ S) {
    int row = (blockIdx.x * 256 + threadIdx.x) >> 5;      // 512 rows
    int lane = threadIdx.x & 31;
    const float* g = G + (size_t)row * 256;
    float v[8]; float mx = -1e30f;
    #pragma unroll
    for (int i = 0; i < 8; ++i) { v[i] = g[lane + i * 32]; mx = fmaxf(mx, v[i]); }
    #pragma unroll
    for (int m = 16; m >= 1; m >>= 1) mx = fmaxf(mx, __shfl_xor(mx, m, 32));
    float sum = 0.f;
    #pragma unroll
    for (int i = 0; i < 8; ++i) { v[i] = __expf(v[i] - mx); sum += v[i]; }
    #pragma unroll
    for (int m = 16; m >= 1; m >>= 1) sum += __shfl_xor(sum, m, 32);
    float inv = 1.f / sum;
    #pragma unroll
    for (int i = 0; i < 8; ++i) S[(size_t)row * 256 + lane + i * 32] = v[i] * inv;
}

// ============================================================================
// 11) out_a[l,e] = sum_d v[l,d] * S[b,e,d]   (wave: 16x64)
// ============================================================================
__global__ void k_av(const unsigned short* __restrict__ vb, const float* __restrict__ S,
                     float* __restrict__ out_a) {
    int wave = (blockIdx.x * 256 + threadIdx.x) >> 5;
    int lane = threadIdx.x & 31, lo = lane & 15, hi = lane >> 4;
    int b = wave >> 10;
    int rem = wave & 1023; int mt = rem >> 2; int ng = rem & 3;
    const unsigned short* arow = vb + ((size_t)b * 4096 + mt * 16 + lo) * 256;
    const float* Sb = S + (size_t)b * 256 * 256;
    floatx8 acc[4] = {zero8(), zero8(), zero8(), zero8()};
    #pragma unroll
    for (int kb = 0; kb < 256; kb += 32) {
        bf16x16 a = load_a_bf16(arow, kb, hi);
        #pragma unroll
        for (int nt = 0; nt < 4; ++nt) {
            int e = ng * 64 + nt * 16 + lo;
            acc[nt] = wmma_bf16(a, load_b_f32(Sb + (size_t)e * 256, kb, hi), acc[nt]);
        }
    }
    size_t mb = (size_t)b * 4096 + mt * 16 + hi * 8;
    #pragma unroll
    for (int nt = 0; nt < 4; ++nt) {
        int e = ng * 64 + nt * 16 + lo;
        #pragma unroll
        for (int r = 0; r < 8; ++r) out_a[(mb + r) * 256 + e] = acc[nt][r];
    }
}

// ============================================================================
// 12) out_a (reinterpreted NCHW) -> NHWC bf16
// ============================================================================
__global__ void k_nhwc(const float* __restrict__ out_a, unsigned short* __restrict__ nhwc1) {
    int id = blockIdx.x;                       // 8192 blocks: b(2) x c(256) x pc(16)
    int b = id >> 12; int c = (id >> 4) & 255; int pc = id & 15;
    int p = pc * 256 + threadIdx.x;
    float v = out_a[(size_t)b * 1048576 + (size_t)c * 4096 + p];
    nhwc1[(size_t)b * 1048576 + (size_t)p * 256 + c] = f2bf_u(v);
}

// ============================================================================
// 13) 3x3 conv, 256->128, implicit GEMM (A=weights m=cout, B=NHWC input n=pos)
//     mode 0: fuse1 -> cat[:, :, 0..127] bf16 ; mode 1: fuse2 -> NCHW f32 + x
// ============================================================================
__global__ void k_conv3x3(const unsigned short* __restrict__ in_nhwc,
                          const unsigned short* __restrict__ wpack,
                          const float* __restrict__ bias, int mode,
                          const float* __restrict__ xres,
                          float* __restrict__ out_f32,
                          unsigned short* __restrict__ out_cat) {
    int wave = (blockIdx.x * 256 + threadIdx.x) >> 5;
    int lane = threadIdx.x & 31, lo = lane & 15, hi = lane >> 4;
    int b = wave >> 11;
    int rem = wave & 2047; int ct = rem >> 8; int pt = rem & 255;
    int p0 = pt * 16; int y0 = p0 >> 6; int x0 = p0 & 63;
    const unsigned short* inb = in_nhwc + (size_t)b * 1048576;
    floatx8 acc = zero8();
    #pragma unroll
    for (int tap = 0; tap < 9; ++tap) {
        int dy = tap / 3 - 1, dx = tap % 3 - 1;
        int yy = y0 + dy;
        if (yy < 0 || yy >= 64) continue;
        int xx = x0 + lo + dx;
        bool valid = (xx >= 0 && xx < 64);
        const unsigned short* brow = inb + (size_t)(yy * 64 + xx) * 256;
        const unsigned short* arow = wpack + (size_t)((ct * 16 + lo) * 9 + tap) * 256;
        #pragma unroll
        for (int kb = 0; kb < 256; kb += 32) {
            bf16x16 a = load_a_bf16(arow, kb, hi);
            bf16x16 bf = valid ? load_b_bf16(brow, kb, hi) : zfrag();
            acc = wmma_bf16(a, bf, acc);
        }
    }
    int p = p0 + lo;
    if (mode == 0) {
        unsigned short* ob = out_cat + (size_t)b * 1048576;
        #pragma unroll
        for (int r = 0; r < 8; ++r) {
            int c = ct * 16 + hi * 8 + r;
            ob[(size_t)p * 256 + c] = f2bf_u(acc[r] + bias[c]);
        }
    } else {
        float* ob = out_f32 + (size_t)b * 128 * 4096;
        const float* xb = xres + (size_t)b * 128 * 4096;
        #pragma unroll
        for (int r = 0; r < 8; ++r) {
            int c = ct * 16 + hi * 8 + r;
            ob[(size_t)c * 4096 + p] = acc[r] + bias[c] + xb[(size_t)c * 4096 + p];
        }
    }
}

// ============================================================================
extern "C" void kernel_launch(void* const* d_in, const int* in_sizes, int n_in,
                              void* d_out, int out_size, void* d_ws, size_t ws_size,
                              hipStream_t stream) {
    const float* x      = (const float*)d_in[0];
    const float* ln_w   = (const float*)d_in[1];
    const float* ln_b   = (const float*)d_in[2];
    const float* ipw    = (const float*)d_in[3];
    const float* cw[3]  = {(const float*)d_in[4],  (const float*)d_in[11], (const float*)d_in[18]};
    const float* cb[3]  = {(const float*)d_in[5],  (const float*)d_in[12], (const float*)d_in[19]};
    const float* xw[3]  = {(const float*)d_in[6],  (const float*)d_in[13], (const float*)d_in[20]};
    const float* dw[3]  = {(const float*)d_in[7],  (const float*)d_in[14], (const float*)d_in[21]};
    const float* db[3]  = {(const float*)d_in[8],  (const float*)d_in[15], (const float*)d_in[22]};
    const float* Al[3]  = {(const float*)d_in[9],  (const float*)d_in[16], (const float*)d_in[23]};
    const float* Dv[3]  = {(const float*)d_in[10], (const float*)d_in[17], (const float*)d_in[24]};
    const float* opw    = (const float*)d_in[25];
    const float* f1w    = (const float*)d_in[26];
    const float* f1b    = (const float*)d_in[27];
    const float* f2w    = (const float*)d_in[28];
    const float* f2b    = (const float*)d_in[29];
    float* out = (float*)d_out;

    // ---- workspace carving (256B aligned) ----
    char* ws = (char*)d_ws;
    size_t off = 0;
    auto alloc = [&](size_t bytes) -> char* {
        char* p = ws + off;
        off = (off + bytes + 255) & ~(size_t)255;
        return p;
    };
    float* xz     = (float*)alloc((size_t)NTOK * 512 * 4);
    float* uc     = (float*)alloc((size_t)3 * NTOK * 256 * 4);   // reused as out_a later
    float* proj   = (float*)alloc((size_t)3 * NTOK * 48 * 4);
    float* delta  = (float*)alloc((size_t)3 * NTOK * 256 * 4);   // reused as nhwc1 later
    float* yscan  = (float*)alloc((size_t)3 * NTOK * 256 * 4);
    float* G      = (float*)alloc((size_t)2 * 256 * 256 * 4);
    float* S      = (float*)alloc((size_t)2 * 256 * 256 * 4);
    unsigned short* xnb   = (unsigned short*)alloc((size_t)NTOK * 128 * 2);
    unsigned short* wip   = (unsigned short*)alloc((size_t)512 * 128 * 2);
    unsigned short* wxp   = (unsigned short*)alloc((size_t)3 * 48 * 256 * 2);
    unsigned short* wop   = (unsigned short*)alloc((size_t)128 * 256 * 2);
    unsigned short* wf1   = (unsigned short*)alloc((size_t)128 * 9 * 256 * 2);
    unsigned short* wf2   = (unsigned short*)alloc((size_t)128 * 9 * 256 * 2);
    unsigned short* ysum  = (unsigned short*)alloc((size_t)NTOK * 256 * 2);
    unsigned short* qb    = (unsigned short*)alloc((size_t)NTOK * 256 * 2);
    unsigned short* kbuf  = (unsigned short*)alloc((size_t)NTOK * 256 * 2);
    unsigned short* vb    = (unsigned short*)alloc((size_t)NTOK * 256 * 2);
    unsigned short* cat   = (unsigned short*)alloc((size_t)NTOK * 256 * 2);
    float* out_a = uc;                       // uc dead after scan
    unsigned short* nhwc1 = (unsigned short*)delta;   // delta dead after scan
    (void)ws_size; (void)n_in; (void)in_sizes; (void)out_size;

    k_pack<<<256, 256, 0, stream>>>(ipw, xw[0], xw[1], xw[2], opw, f1w, f2w,
                                    wip, wxp, wop, wf1, wf2);
    k_ln<<<NTOK, 128, 0, stream>>>(x, ln_w, ln_b, xnb);
    k_inproj<<<512, 256, 0, stream>>>(xnb, wip, xz);
    k_conv1d<<<dim3(NTOK, 3), 256, 0, stream>>>(xz, cw[0], cb[0], cw[1], cb[1], cw[2], cb[2], uc);
    k_xproj<<<dim3(64, 3), 256, 0, stream>>>(uc, wxp, proj);
    k_delta<<<dim3(NTOK, 3), 256, 0, stream>>>(proj, dw[0], db[0], dw[1], db[1], dw[2], db[2], delta);
    k_scan<<<96, 256, 0, stream>>>(uc, proj, delta, Al[0], Dv[0], Al[1], Dv[1], Al[2], Dv[2], yscan);
    k_combine<<<NTOK, 256, 0, stream>>>(yscan, xz, ysum, qb, kbuf, vb);
    k_outproj<<<128, 256, 0, stream>>>(ysum, wop, cat);
    k_qk<<<32, 256, 0, stream>>>(qb, kbuf, G);
    k_softmax<<<64, 256, 0, stream>>>(G, S);
    k_av<<<256, 256, 0, stream>>>(vb, S, out_a);
    k_nhwc<<<8192, 256, 0, stream>>>(out_a, nhwc1);
    k_conv3x3<<<512, 256, 0, stream>>>(nhwc1, wf1, f1b, 0, nullptr, nullptr, cat);
    k_conv3x3<<<512, 256, 0, stream>>>(cat, wf2, f2b, 1, x, out, nullptr);
}
